// GridCLIPModel_15917148799094
// MI455X (gfx1250) — matrix operand order repaired and checked by reading, not compiled, not run on hardware
//
#include <hip/hip_runtime.h>
#include <hip/hip_bf16.h>

typedef __attribute__((ext_vector_type(8)))  _Float16 v8h;
typedef __attribute__((ext_vector_type(16))) _Float16 v16h;
typedef __attribute__((ext_vector_type(8)))  float    v8f;

#define NPOINTS        524288
#define ROWS_PER_BLOCK 64
#define THREADS        128
#define CLIPD          512

// f16 transposed-weight offsets (in halves) inside d_ws
#define WT0_OFF 0        // w0t  [256][32]
#define WT1_OFF 8192     // w1t  [256][256]
#define WT2_OFF 73728    // w2t  [1024][256]
#define WT3_OFF 335872   // iw0t [512][512]
#define WT4_OFF 598016   // iw1t [512][512]

// ---- replicated _level_meta() ----
__constant__ float    kScale[16]  = {15.f,31.f,63.f,127.f,255.f,511.f,1023.f,2047.f,
                                     4095.f,8191.f,16383.f,32767.f,65535.f,131071.f,262143.f,524287.f};
__constant__ unsigned kHsize[16]  = {4920u,35944u,65536u,65536u,65536u,65536u,65536u,65536u,
                                     65536u,65536u,65536u,65536u,65536u,65536u,65536u,65536u};
__constant__ unsigned kOffset[16] = {0u,4920u,40864u,106400u,171936u,237472u,303008u,368544u,
                                     434080u,499616u,565152u,630688u,696224u,761760u,827296u,892832u};
__constant__ unsigned kR1[16]     = {17u,33u,0u,0u,0u,0u,0u,0u,0u,0u,0u,0u,0u,0u,0u,0u};
__constant__ int      kUseHash[16]= {0,0,1,1,1,1,1,1,1,1,1,1,1,1,1,1};

// ---------- weight convert: fp32 W[K][N] -> f16 Wt[N][K] ----------
__global__ void wcvt_kernel(const float* __restrict__ w, _Float16* __restrict__ out,
                            int K, int N) {
  int i = blockIdx.x * blockDim.x + threadIdx.x;
  if (i >= K * N) return;
  int k = i / N;
  int n = i - k * N;
  out[(size_t)n * K + k] = (_Float16)w[i];
}

// ---------- WMMA fragment loaders (CDNA5 16x16x32 f16 layouts, wave32) ----------
// A (16x32, MxK): lane group g, row M=lane&15; K = {g*8+e, e<8} U {16+g*8+e}
__device__ __forceinline__ v16h load_a_frag(const _Float16* in, int ld, int row0,
                                            int kk, int lane) {
  const int g = lane >> 4, m = lane & 15;
  const _Float16* rp = in + (row0 + m) * ld + (kk << 5);
  v8h lo = *(const v8h*)(rp + g * 8);
  v8h hi = *(const v8h*)(rp + 16 + g * 8);
  v16h a;
#pragma unroll
  for (int i = 0; i < 8; ++i) { a[i] = lo[i]; a[8 + i] = hi[i]; }
  return a;
}

// B (32x16, KxN): lane group g holds K = {g*8+e, e<8} U {16+g*8+e}, N = lane&15.
// Weights stored transposed Wt[n][k] -> two runs of 8 contiguous halves per lane.
__device__ __forceinline__ v16h load_b_frag(const _Float16* __restrict__ wt, int K,
                                            int nbase, int kk, int lane) {
  const int g = lane >> 4, n = lane & 15;
  const _Float16* p = wt + (size_t)(nbase + n) * K + (kk << 5) + g * 8;
  v8h lo = *(const v8h*)(p);
  v8h hi = *(const v8h*)(p + 16);
  v16h b;
#pragma unroll
  for (int i = 0; i < 8; ++i) { b[i] = lo[i]; b[8 + i] = hi[i]; }
  return b;
}

// ---------- block GEMM: waves partition N; each wave: 4 N-tiles x 4 M-tiles ------
// out = [relu](in @ W + b); in: LDS f16 [64][ldin]; W: global f16 Wt[n][K]
// Per K-step: 4 B fragment loads + 4 A fragment loads -> 16 WMMAs.
template <bool RELU, bool TO_GLOBAL>
__device__ __forceinline__ void block_gemm(
    const _Float16* in, int ldin, int K,
    const _Float16* __restrict__ wt, const float* __restrict__ bias, int NOUT,
    _Float16* outl, int ldoutl, float* __restrict__ outg, int ldoutg,
    int wave, int lane) {
  const int g = lane >> 4, n = lane & 15;
  const int per_wave = (NOUT >> 4) >> 2;              // n-tiles per wave: 4 or 8
  const int nt_begin = wave * per_wave * 16;
  const int nt_end   = nt_begin + per_wave * 16;
  for (int nt = nt_begin; nt < nt_end; nt += 64) {    // four 16-col tiles at a time
    v8f acc[4][4];                                    // [n-subtile][m-tile]
#pragma unroll
    for (int j = 0; j < 4; ++j) {
      const float bv = bias[nt + j * 16 + n];
#pragma unroll
      for (int mt = 0; mt < 4; ++mt)
#pragma unroll
        for (int v = 0; v < 8; ++v) acc[j][mt][v] = bv;
    }
    const int ksteps = K >> 5;
    for (int kk = 0; kk < ksteps; ++kk) {
      v16h b[4];
#pragma unroll
      for (int j = 0; j < 4; ++j) b[j] = load_b_frag(wt, K, nt + j * 16, kk, lane);
#pragma unroll
      for (int mt = 0; mt < 4; ++mt) {
        const v16h a = load_a_frag(in, ldin, mt * 16, kk, lane);
#pragma unroll
        for (int j = 0; j < 4; ++j)
          acc[j][mt] = __builtin_amdgcn_wmma_f32_16x16x32_f16(
              false, a, false, b[j], (short)0, acc[j][mt], false, false);
      }
    }
#pragma unroll
    for (int mt = 0; mt < 4; ++mt) {
#pragma unroll
      for (int j = 0; j < 4; ++j) {
#pragma unroll
        for (int v = 0; v < 8; ++v) {
          float r = acc[j][mt][v];
          if (RELU) r = r > 0.0f ? r : 0.0f;
          const int row = mt * 16 + g * 8 + v;
          const int col = nt + j * 16 + n;
          if (TO_GLOBAL) outg[(size_t)row * ldoutg + col] = r;
          else           outl[row * ldoutl + col] = (_Float16)r;
        }
      }
    }
  }
}

// ---------- fully fused: grid-encode + 5-layer MLP per 64-row tile ----------
__global__ __launch_bounds__(THREADS, 1) void gridclip_fused(
    const float* __restrict__ x, const float* __restrict__ emb,
    const _Float16* __restrict__ wt,
    const float* __restrict__ b0, const float* __restrict__ b1,
    const float* __restrict__ b2, const float* __restrict__ ib0,
    const float* __restrict__ ib1,
    float* __restrict__ out_label, float* __restrict__ out_image) {
  extern __shared__ _Float16 smem[];
  _Float16* actA = smem;                              // 64 x 512 halves
  _Float16* actB = smem + ROWS_PER_BLOCK * 512;       // 64 x 512 halves

  const int tid  = threadIdx.x;
  const int lane = tid & 31;
  const int wave = tid >> 5;
  const int rbase = blockIdx.x * ROWS_PER_BLOCK;

  __builtin_prefetch(wt + WT0_OFF, 0, 1);

  // ---- stage this block's coordinates into LDS (actB region is dead here) ----
  float* xs = (float*)actB;                           // 64*3 floats
  for (int i = tid; i < ROWS_PER_BLOCK * 3; i += THREADS)
    xs[i] = x[(size_t)rbase * 3 + i];
  __syncthreads();

  // ---- multi-level hash-grid encode -> actA[row][32] (f16) ----
  for (int p = tid; p < ROWS_PER_BLOCK * 16; p += THREADS) {
    const int row = p >> 4, lvl = p & 15;
    const float u0 = (xs[row * 3 + 0] + 11.0f) * (1.0f / 22.0f);
    const float u1 = (xs[row * 3 + 1] + 11.0f) * (1.0f / 22.0f);
    const float u2 = (xs[row * 3 + 2] + 11.0f) * (1.0f / 22.0f);
    const float s = kScale[lvl];
    const float p0 = u0 * s + 0.5f, p1 = u1 * s + 0.5f, p2 = u2 * s + 0.5f;
    const float g0 = floorf(p0), g1 = floorf(p1), g2 = floorf(p2);
    const float f0 = p0 - g0, f1 = p1 - g1, f2 = p2 - g2;
    const unsigned c0 = (unsigned)g0, c1 = (unsigned)g1, c2 = (unsigned)g2;
    const unsigned hs = kHsize[lvl], off = kOffset[lvl];
    const int      uh = kUseHash[lvl];
    const unsigned r1 = kR1[lvl];
    float a0 = 0.0f, a1 = 0.0f;
#pragma unroll
    for (int corner = 0; corner < 8; ++corner) {
      const unsigned d0 = corner & 1, d1 = (corner >> 1) & 1, d2 = (corner >> 2) & 1;
      const unsigned cc0 = c0 + d0, cc1 = c1 + d1, cc2 = c2 + d2;
      const float w = (d0 ? f0 : 1.0f - f0) * (d1 ? f1 : 1.0f - f1) *
                      (d2 ? f2 : 1.0f - f2);
      unsigned idx;
      if (uh) idx = cc0 ^ (cc1 * 2654435761u) ^ (cc2 * 805459861u);
      else    idx = cc0 + cc1 * r1 + cc2 * r1 * r1;
      idx = idx % hs;
      const float2 gv = *(const float2*)(emb + 2ull * (off + idx));
      a0 += w * gv.x;
      a1 += w * gv.y;
    }
    actA[row * 32 + lvl * 2]     = (_Float16)a0;
    actA[row * 32 + lvl * 2 + 1] = (_Float16)a1;
  }
  __builtin_prefetch(wt + WT1_OFF, 0, 1);
  __syncthreads();

  // L1: [64x32] @ w0 -> relu -> actB [64x256]
  block_gemm<true, false>(actA, 32, 32, wt + WT0_OFF, b0, 256,
                          actB, 256, (float*)nullptr, 0, wave, lane);
  __builtin_prefetch(wt + WT2_OFF, 0, 1);
  __syncthreads();
  // L2: [64x256] @ w1 -> relu -> actA [64x256]
  block_gemm<true, false>(actB, 256, 256, wt + WT1_OFF, b1, 256,
                          actA, 256, (float*)nullptr, 0, wave, lane);
  __builtin_prefetch(wt + WT3_OFF, 0, 1);
  __syncthreads();
  // L3a: label half -> global fp32
  block_gemm<false, true>(actA, 256, 256, wt + WT2_OFF, b2, 512,
                          (_Float16*)nullptr, 0,
                          out_label + (size_t)rbase * CLIPD, CLIPD, wave, lane);
  // L3b: image half (no relu) -> actB [64x512]
  block_gemm<false, false>(actA, 256, 256, wt + WT2_OFF + (size_t)512 * 256,
                           b2 + 512, 512, actB, 512, (float*)nullptr, 0, wave, lane);
  __builtin_prefetch(wt + WT4_OFF, 0, 1);
  __syncthreads();
  // L4: relu(I0 @ iw0 + ib0) -> actA [64x512]
  block_gemm<true, false>(actB, 512, 512, wt + WT3_OFF, ib0, 512,
                          actA, 512, (float*)nullptr, 0, wave, lane);
  __syncthreads();
  // L5: I1 @ iw1 + ib1 -> global fp32
  block_gemm<false, true>(actA, 512, 512, wt + WT4_OFF, ib1, 512,
                          (_Float16*)nullptr, 0,
                          out_image + (size_t)rbase * CLIPD, CLIPD, wave, lane);
}

extern "C" void kernel_launch(void* const* d_in, const int* in_sizes, int n_in,
                              void* d_out, int out_size, void* d_ws, size_t ws_size,
                              hipStream_t stream) {
  (void)in_sizes; (void)n_in; (void)out_size; (void)ws_size;
  const float* x   = (const float*)d_in[0];
  const float* emb = (const float*)d_in[1];
  const float* w0  = (const float*)d_in[2];
  const float* b0  = (const float*)d_in[3];
  const float* w1  = (const float*)d_in[4];
  const float* b1  = (const float*)d_in[5];
  const float* w2  = (const float*)d_in[6];
  const float* b2  = (const float*)d_in[7];
  const float* iw0 = (const float*)d_in[8];
  const float* ib0 = (const float*)d_in[9];
  const float* iw1 = (const float*)d_in[10];
  const float* ib1 = (const float*)d_in[11];

  _Float16* wt = (_Float16*)d_ws;
  float* out_label = (float*)d_out;
  float* out_image = out_label + (size_t)NPOINTS * CLIPD;

  // weights -> f16, transposed (Wt[n][k])
  wcvt_kernel<<<(32 * 256 + 255) / 256,   256, 0, stream>>>(w0,  wt + WT0_OFF, 32, 256);
  wcvt_kernel<<<(256 * 256 + 255) / 256,  256, 0, stream>>>(w1,  wt + WT1_OFF, 256, 256);
  wcvt_kernel<<<(256 * 1024 + 255) / 256, 256, 0, stream>>>(w2,  wt + WT2_OFF, 256, 1024);
  wcvt_kernel<<<(512 * 512 + 255) / 256,  256, 0, stream>>>(iw0, wt + WT3_OFF, 512, 512);
  wcvt_kernel<<<(512 * 512 + 255) / 256,  256, 0, stream>>>(iw1, wt + WT4_OFF, 512, 512);

  const int lds_bytes = ROWS_PER_BLOCK * 512 * 2 * (int)sizeof(_Float16); // 128 KB
  (void)hipFuncSetAttribute((const void*)gridclip_fused,
                            hipFuncAttributeMaxDynamicSharedMemorySize, lds_bytes);
  gridclip_fused<<<NPOINTS / ROWS_PER_BLOCK, THREADS, lds_bytes, stream>>>(
      x, emb, wt, b0, b1, b2, ib0, ib1, out_label, out_image);
}
